// ElectronicSchNet_67207648247821
// MI455X (gfx1250) — matrix operand order repaired and checked by reading, not compiled, not run on hardware
//
#include <hip/hip_runtime.h>

typedef __attribute__((ext_vector_type(16))) _Float16 v16h;
typedef __attribute__((ext_vector_type(8)))  _Float16 v8h;
typedef __attribute__((ext_vector_type(8)))  float    v8f;

#define NE      16      // electrons
#define ED      128     // EDIM
#define KD      64      // KDIM
#define DF      32      // DFEAT
#define NL      3       // layers
#define TPB     128     // 4 wave32 per block

// edge groups (all multiples of 16 -> perfect WMMA tiles)
#define N_SAME  112     // uu(56)+dd(56)
#define N_ANTI  128     // ud(64)+du(64)
#define N_NUC   64      // 16 elec x 4 nuc
#define NEDGE   304
#define RB_ANTI 112
#define RB_NUC  240
#define NTILES  19      // 7 same + 8 anti + 4 nuc

// packed f16 weight layout in d_ws (element offsets)
#define OFF_HW  0
#define SZ_HW   (3*128*64)          // hW, 4 ksteps x 4 ntiles per layer
#define OFF_W0  (OFF_HW + SZ_HW)
#define SZ_W0   (9*32*64)           // wW0 padded N 45->64
#define OFF_W1  (OFF_W0 + SZ_W0)
#define SZ_W1   (9*64*64)           // wW1 padded K 45->64
#define OFF_G   (OFF_W1 + SZ_W1)
#define SZ_G    (9*64*128)
#define PACKED_TOTAL (OFF_G + SZ_G) // 153600 f16 = 300 KB of d_ws

// ---- edge index decode (matches reference idx_perm ordering) ----
__device__ __forceinline__ void same_ij(int e, int& i, int& j) {
  if (e < 56) { i = e / 7; int jj = e % 7; j = jj + (jj >= i); }
  else { e -= 56; int ii = e / 7; int jj = e % 7; i = 8 + ii; j = 8 + jj + (jj >= ii); }
}
__device__ __forceinline__ void anti_ij(int e, int& i, int& j) {
  if (e < 64) { i = e / 8; j = 8 + (e & 7); }
  else { e -= 64; i = 8 + (e / 8); j = e & 7; }
}

// A fragment (16x32 f16, row-major LDS, row stride ld):
// lane L: row = L&15; elems 0..7 -> K = kb..kb+7, elems 8..15 -> K = kb+16..kb+23, kb = 8*(L>>4)
__device__ __forceinline__ v16h ld_afrag(const _Float16* base, int ld, int lane) {
  const int row = lane & 15;
  const int kb  = (lane >> 4) << 3;
  const _Float16* p = base + row * ld + kb;
  v8h lo = *reinterpret_cast<const v8h*>(p);
  v8h hi = *reinterpret_cast<const v8h*>(p + 16);
  v16h a;
#pragma unroll
  for (int e = 0; e < 8; ++e) { a[e] = lo[e]; a[e + 8] = hi[e]; }
  return a;
}

// B fragment from pre-swizzled global weights: 16 contiguous f16 per (frag, lane)
__device__ __forceinline__ v16h ld_bfrag(const _Float16* packed, int frag, int lane) {
  const _Float16* p = packed + (size_t)((frag << 5) + lane) * 16;
  v8h lo = *reinterpret_cast<const v8h*>(p);
  v8h hi = *reinterpret_cast<const v8h*>(p + 8);
  v16h b;
#pragma unroll
  for (int e = 0; e < 8; ++e) { b[e] = lo[e]; b[e + 8] = hi[e]; }
  return b;
}

__device__ __forceinline__ v8f wmma_f16(v16h a, v16h b, v8f c) {
  return __builtin_amdgcn_wmma_f32_16x16x32_f16(false, a, false, b, (short)0, c, false, false);
}

// shifted softplus: logaddexp(x,0) - ln2 = max(x,0) + log(1+exp(-|x|)) - ln2
// fast path: hardware v_exp/v_log transcendentals (pipeline is f16 precision anyway)
__device__ __forceinline__ float ssp(float x) {
  return fmaxf(x, 0.0f) + __logf(1.0f + __expf(-fabsf(x))) - 0.69314718056f;
}

// ---- kernel 1: convert all weights to f16, zero-pad, swizzle to fragment-major B layout ----
// B frag covers K rows [s*32, s*32+32), cols [t*16, t*16+16):
//   dst[((s*NT + t)*32 + L)*16 + e] = src[(s*32 + (e&7) + 8*(L>>4) + 16*(e>>3)) * Nsrc + t*16 + (L&15)]  (0 if OOB)
__global__ void prep_weights(const float* __restrict__ wW0, const float* __restrict__ wW1,
                             const float* __restrict__ hW, const float* __restrict__ gW,
                             _Float16* __restrict__ packed) {
  for (int p = blockIdx.x * blockDim.x + threadIdx.x; p < PACKED_TOTAL;
       p += gridDim.x * blockDim.x) {
    const float* src; int Ksrc, Nsrc, NT, local;
    if (p < OFF_W0)      { int l = p / 8192;              local = p % 8192;           src = hW  + l * 8192;      Ksrc = 128; Nsrc = 64;  NT = 4; }
    else if (p < OFF_W1) { int q = p - OFF_W0; int m = q / 2048; local = q % 2048;    src = wW0 + m * (32 * 45); Ksrc = 32;  Nsrc = 45;  NT = 4; }
    else if (p < OFF_G)  { int q = p - OFF_W1; int m = q / 4096; local = q % 4096;    src = wW1 + m * (45 * 64); Ksrc = 45;  Nsrc = 64;  NT = 4; }
    else                 { int q = p - OFF_G;  int m = q / 8192; local = q % 8192;    src = gW  + m * (64 * 128);Ksrc = 64;  Nsrc = 128; NT = 8; }
    int e = local & 15, L = (local >> 4) & 31, ft = local >> 9;
    int t = ft % NT, s = ft / NT;
    int k = s * 32 + (e & 7) + ((L >> 4) << 3) + ((e >> 3) << 4);
    int c = t * 16 + (L & 15);
    float v = (k < Ksrc && c < Nsrc) ? src[k * Nsrc + c] : 0.0f;
    packed[p] = (_Float16)v;
  }
}

// ---- kernel 2: one workgroup (4 wave32) per batch element, fully fused 3 layers ----
__global__ __launch_bounds__(TPB)
void schnet_fused(const float* __restrict__ de, const float* __restrict__ dn,
                  const float* __restrict__ X,  const float* __restrict__ Yg,
                  const float* __restrict__ wb0, const _Float16* __restrict__ packed,
                  float* __restrict__ out) {
  __shared__ alignas(16) float    xbuf[2][NE * ED];   // 16 KB  ping-pong x
  __shared__ alignas(16) _Float16 xh[NE * ED];        //  4 KB  f16 copy of x for WMMA A
  __shared__ alignas(16) _Float16 Efeat[NEDGE][DF];   // 19 KB  edge features (layer-invariant)
  __shared__ alignas(16) float    hmat[NE][KD];       //  4 KB  h = x @ hW
  __shared__ alignas(16) _Float16 wmat[NEDGE][KD];    // 38 KB  per-edge subnet outputs
  __shared__ alignas(16) _Float16 zh[3][NE][KD];      //  6 KB  reduced messages (f16 A for gW GEMM)
  __shared__ alignas(16) _Float16 act[4][NE][KD];     //  8 KB  per-wave ssp activations
  __shared__ alignas(16) float    Ysh[4][KD];         //  1 KB

  const int b    = blockIdx.x;
  const int tid  = threadIdx.x;
  const int lane = tid & 31;
  const int wv   = tid >> 5;

  // ---- per-block one-time setup ----
  for (int idx = tid; idx < NE * ED; idx += TPB) xbuf[0][idx] = X[idx % ED];
  for (int idx = tid; idx < 4 * KD;  idx += TPB) Ysh[idx / KD][idx % KD] = Yg[idx];
  // dist_basis (nocusp) into tile-ordered f16 edge matrix
  for (int idx = tid; idx < NEDGE * DF; idx += TPB) {
    int edge = idx / DF, f = idx % DF;
    float d;
    if (edge < N_SAME)      { int i, j; same_ij(edge, i, j);            d = de[b * 256 + i * 16 + j]; }
    else if (edge < RB_NUC) { int i, j; anti_ij(edge - RB_ANTI, i, j);  d = de[b * 256 + i * 16 + j]; }
    else                    { d = dn[b * 64 + (edge - RB_NUC)]; }       // row = i*4+m matches dn layout
    float q   = (float)(2 * f + 1) * (1.0f / 64.0f);
    float mu  = 10.0f * q * q;
    float sg  = (1.0f + 10.0f * q) * (1.0f / 7.0f);
    float env = d * d * __expf(-d);
    float tdm = (d - mu) / sg;
    Efeat[edge][f] = (_Float16)(env * __expf(-tdm * tdm));
  }
  __syncthreads();

  int cur = 0;
#pragma unroll 1
  for (int l = 0; l < NL; ++l) {
    for (int idx = tid; idx < NE * ED; idx += TPB) xh[idx] = (_Float16)xbuf[cur][idx];
    __syncthreads();

    // ---- h = x @ hW[l] : M=16, N=64 (one 16-col tile per wave), K=128 ----
    {
      const _Float16* HW = packed + OFF_HW + l * 8192;
      v8f acc = {};
#pragma unroll
      for (int s = 0; s < 4; ++s)
        acc = wmma_f16(ld_afrag(xh + s * 32, ED, lane), ld_bfrag(HW, s * 4 + wv, lane), acc);
      const int col = (lane & 15) + 16 * wv;
      const int rb  = (lane >> 4) << 3;
#pragma unroll
      for (int v = 0; v < 8; ++v) hmat[rb + v][col] = acc[v];
    }

    // ---- w-subnet over 19 edge tiles, round-robin across waves ----
    for (int tt = wv; tt < NTILES; tt += 4) {
      int kind, rowbase;
      if (tt < 7)       { kind = 0; rowbase = tt * 16; }
      else if (tt < 15) { kind = 1; rowbase = RB_ANTI + (tt - 7) * 16; }
      else              { kind = 2; rowbase = RB_NUC  + (tt - 15) * 16; }
      const _Float16* W0 = packed + OFF_W0 + (l * 3 + kind) * 2048;
      const _Float16* W1 = packed + OFF_W1 + (l * 3 + kind) * 4096;
      const float*    bs = wb0 + (l * 3 + kind) * 45;
      const int rb = (lane >> 4) << 3;

      // GEMM1 (16x32 @ 32x64) + bias + ssp -> act (padded cols give ssp(0)=0, killed by zero W1 rows)
      v16h ea = ld_afrag(&Efeat[rowbase][0], DF, lane);
#pragma unroll
      for (int nt = 0; nt < 4; ++nt) {
        v8f c = {};
        c = wmma_f16(ea, ld_bfrag(W0, nt, lane), c);
        const int col = (lane & 15) + 16 * nt;
        const float bv = (col < 45) ? bs[col] : 0.0f;
#pragma unroll
        for (int v = 0; v < 8; ++v) act[wv][rb + v][col] = (_Float16)ssp(c[v] + bv);
      }
      // same-wave DS RAW: act stores above -> act fragment loads below (DS in-order per wave)
      asm volatile("s_wait_dscnt 0" ::: "memory");
      // GEMM2 (16x64 @ 64x64) -> per-edge w values
#pragma unroll
      for (int nt = 0; nt < 4; ++nt) {
        v8f c = {};
#pragma unroll
        for (int s = 0; s < 2; ++s)
          c = wmma_f16(ld_afrag(&act[wv][0][0] + s * 32, KD, lane),
                       ld_bfrag(W1, s * 4 + nt, lane), c);
        const int col = (lane & 15) + 16 * nt;
#pragma unroll
        for (int v = 0; v < 8; ++v) wmat[rowbase + rb + v][col] = (_Float16)c[v];
      }
    }
    __syncthreads();  // hmat + wmat complete

    // ---- deterministic fixed-order neighbor reduction: z[kind][i][k] ----
    for (int idx = tid; idx < 3 * NE * KD; idx += TPB) {
      int kind = idx >> 10, rem = idx & 1023, i = rem >> 6, k = rem & 63;
      float sum = 0.0f;
      if (kind == 0) {
        if (i < 8) {
#pragma unroll
          for (int jj = 0; jj < 7; ++jj) { int j = jj + (jj >= i);
            sum += (float)wmat[i * 7 + jj][k] * hmat[j][k]; }
        } else { int ii = i - 8;
#pragma unroll
          for (int jj = 0; jj < 7; ++jj) { int j = 8 + jj + (jj >= ii);
            sum += (float)wmat[56 + ii * 7 + jj][k] * hmat[j][k]; }
        }
      } else if (kind == 1) {
        if (i < 8) {
#pragma unroll
          for (int jj = 0; jj < 8; ++jj) sum += (float)wmat[RB_ANTI + i * 8 + jj][k] * hmat[8 + jj][k];
        } else {
#pragma unroll
          for (int jj = 0; jj < 8; ++jj) sum += (float)wmat[RB_ANTI + 64 + (i - 8) * 8 + jj][k] * hmat[jj][k];
        }
      } else {
#pragma unroll
        for (int m = 0; m < 4; ++m) sum += (float)wmat[RB_NUC + i * 4 + m][k] * Ysh[m][k];
      }
      zh[kind][i][k] = (_Float16)sum;
    }
    __syncthreads();

    // ---- x_next = x + sum_kind z_kind @ gW[l][kind] : M=16, N=128 (2 col tiles/wave), K=64 ----
    for (int nt = wv; nt < 8; nt += 4) {
      const int col = (lane & 15) + 16 * nt;
      const int rb  = (lane >> 4) << 3;
      v8f c;
#pragma unroll
      for (int v = 0; v < 8; ++v) c[v] = xbuf[cur][(rb + v) * ED + col];
#pragma unroll
      for (int kind = 0; kind < 3; ++kind) {
        const _Float16* G = packed + OFF_G + (l * 3 + kind) * 8192;
#pragma unroll
        for (int s = 0; s < 2; ++s)
          c = wmma_f16(ld_afrag(&zh[kind][0][0] + s * 32, KD, lane),
                       ld_bfrag(G, s * 8 + nt, lane), c);
      }
#pragma unroll
      for (int v = 0; v < 8; ++v) xbuf[cur ^ 1][(rb + v) * ED + col] = c[v];
    }
    __syncthreads();
    cur ^= 1;
  }

  for (int idx = tid; idx < NE * ED; idx += TPB)
    out[(size_t)b * (NE * ED) + idx] = xbuf[cur][idx];
}

extern "C" void kernel_launch(void* const* d_in, const int* in_sizes, int n_in,
                              void* d_out, int out_size, void* d_ws, size_t ws_size,
                              hipStream_t stream) {
  (void)n_in; (void)out_size; (void)ws_size;
  const float* de  = (const float*)d_in[0];
  const float* dn  = (const float*)d_in[1];
  const float* X   = (const float*)d_in[2];
  const float* Y   = (const float*)d_in[3];
  const float* wW0 = (const float*)d_in[4];
  const float* wb0 = (const float*)d_in[5];
  const float* wW1 = (const float*)d_in[6];
  const float* hW  = (const float*)d_in[7];
  const float* gW  = (const float*)d_in[8];
  _Float16* packed = (_Float16*)d_ws;     // 300 KB scratch

  prep_weights<<<600, 256, 0, stream>>>(wW0, wW1, hW, gW, packed);

  const int nbatch = in_sizes[0] / (NE * NE);   // dists_elec is [B,16,16]
  schnet_fused<<<nbatch, TPB, 0, stream>>>(de, dn, X, Y, wb0, packed, (float*)d_out);
}
